// GCN_2l_28690381537663
// MI455X (gfx1250) — matrix operand order repaired and checked by reading, not compiled, not run on hardware
//
#include <hip/hip_runtime.h>
#include <math.h>

#define NN 20000
#define EE 320000
#define NHEAD 4
#define EPS_F 1e-16f

typedef __attribute__((ext_vector_type(16))) _Float16 v16h;
typedef __attribute__((ext_vector_type(8)))  float    v8f;
typedef __attribute__((ext_vector_type(4)))  int      v4i;

// --------------------------------------------------------------------------
// CDNA5 async global->LDS path (guarded: falls back to float4 copy).
// Probe round 2 showed this toolchain declares the builtin with generic
// int4* parameters: (v4i* gsrc, v4i* ldst, imm offset, imm cpol).
// --------------------------------------------------------------------------
#if __has_builtin(__builtin_amdgcn_global_load_async_to_lds_b128)
#define HAVE_ASYNC_LDS 1
__device__ __forceinline__ void async_b128(const void* g, void* l) {
    __builtin_amdgcn_global_load_async_to_lds_b128((v4i*)g, (v4i*)l, 0, 0);
}
__device__ __forceinline__ void async_wait0() {
#if __has_builtin(__builtin_amdgcn_s_wait_asynccnt)
    __builtin_amdgcn_s_wait_asynccnt(0);
#else
    asm volatile("s_wait_asynccnt 0" ::: "memory");
#endif
}
#else
#define HAVE_ASYNC_LDS 0
#endif

// --------------------------------------------------------------------------
// Pre-pack X into A-fragments (f16, WMMA register order).
// Apk[((t*4+kk)*32 + lane)*16 + h], t = row tile, kk = K step.
// A 16x32 f16 layout: M = lane&15, K = (h&7) | (((h>>3)<<1 | lane>>4)<<3)
// --------------------------------------------------------------------------
__global__ void pack_a(const float* __restrict__ X, _Float16* __restrict__ Apk)
{
    int i = blockIdx.x * blockDim.x + threadIdx.x;   // NN*128 fragment halves
    if (i >= NN * 128) return;
    int t  = i >> 11;
    int r  = i & 2047;
    int kk = r >> 9;
    int l  = (r >> 4) & 31;
    int h  = r & 15;
    int M  = t * 16 + (l & 15);
    int Ka = kk * 32 + ((h & 7) | ((((h >> 3) << 1) | (l >> 4)) << 3));
    Apk[i] = (_Float16)X[(size_t)M * 128 + Ka];
}

// --------------------------------------------------------------------------
// Pre-pack the 4 layer-1 weights into B-fragments (f16, WMMA order).
// Bpk[mat][((c*4+kk)*32 + lane)*16 + h]; N = c*16 + (lane&15),
// K = kk*32 + (h&7) + 8*(lane>>4) + 16*(h>>3)  (same K order as the A fragments)
// --------------------------------------------------------------------------
__global__ void pack_b(const float* __restrict__ Wq, const float* __restrict__ Wk,
                       const float* __restrict__ Wv, const float* __restrict__ Ws,
                       _Float16* __restrict__ Bpk)
{
    int i = blockIdx.x * blockDim.x + threadIdx.x;   // 4 * 16384
    if (i >= 4 * 16384) return;
    int mat = i >> 14;
    int r   = i & 16383;
    int c   = r >> 11;
    int r2  = r & 2047;
    int kk  = r2 >> 9;
    int l   = (r2 >> 4) & 31;
    int h   = r2 & 15;
    int Nc  = c * 16 + (l & 15);
    int Kb  = kk * 32 + (h & 7) + 8 * (l >> 4) + 16 * (h >> 3);
    const float* W = mat == 0 ? Wq : mat == 1 ? Wk : mat == 2 ? Wv : Ws;
    Bpk[i] = (_Float16)W[Kb * 128 + Nc];
}

// --------------------------------------------------------------------------
// Layer-1 GEMM from pre-packed fragments.
// grid = (1250, 4): x = 16-row tile, y = which matrix. block = 128 (4 waves).
// Each wave computes two 16x16 col tiles sharing one A fragment per K step:
// inner loop = ds_load_b128 (A) + global_load_b128 (B) + 2x v_wmma.
// A tile (4 KB) staged via GLOBAL_LOAD_ASYNC_TO_LDS_B128 when available.
// --------------------------------------------------------------------------
__global__ __launch_bounds__(128)
void gemm_l1(const _Float16* __restrict__ Apk, const _Float16* __restrict__ Bpk,
             const float* __restrict__ ba, const float* __restrict__ bb,
             const float* __restrict__ bc, const float* __restrict__ bd,
             float* __restrict__ Oa, float* __restrict__ Ob,
             float* __restrict__ Oc, float* __restrict__ Od)
{
    const float* b; float* O;
    switch (blockIdx.y) {
        case 0:  b = ba; O = Oa; break;
        case 1:  b = bb; O = Ob; break;
        case 2:  b = bc; O = Oc; break;
        default: b = bd; O = Od; break;
    }
    const _Float16* Bp = Bpk + (size_t)blockIdx.y * 16384;

    __shared__ __align__(32) _Float16 sA[2048];      // 4 KB packed A tile

    const int tid = threadIdx.x;
    const int t   = blockIdx.x;

    const float4* gA = (const float4*)(Apk + (size_t)t * 2048);
#if HAVE_ASYNC_LDS
    #pragma unroll
    for (int i = tid; i < 256; i += 128)
        async_b128((const char*)gA + i * 16, (char*)sA + i * 16);
    async_wait0();
    __syncthreads();
#else
    #pragma unroll
    for (int i = tid; i < 256; i += 128)
        ((float4*)sA)[i] = gA[i];
    __syncthreads();
#endif

    const int wave = tid >> 5;
    const int lane = tid & 31;
    const int ln   = lane & 15;
    const int lh   = lane >> 4;
    const int c0   = wave * 2;
    const int c1   = wave * 2 + 1;

    v8f acc0 = {}, acc1 = {};
    #pragma unroll
    for (int kk = 0; kk < 4; ++kk) {
        v16h a  = *(const v16h*)(sA + (kk * 32 + lane) * 16);
        v16h b0 = *(const v16h*)(Bp + (size_t)((c0 * 4 + kk) * 32 + lane) * 16);
        v16h b1 = *(const v16h*)(Bp + (size_t)((c1 * 4 + kk) * 32 + lane) * 16);
        acc0 = __builtin_amdgcn_wmma_f32_16x16x32_f16(
            false, a, false, b0, (short)0, acc0, false, false);
        acc1 = __builtin_amdgcn_wmma_f32_16x16x32_f16(
            false, a, false, b1, (short)0, acc1, false, false);
    }

    const int col0 = c0 * 16 + ln;
    const int col1 = c1 * 16 + ln;
    const float bv0 = b[col0], bv1 = b[col1];
    #pragma unroll
    for (int r = 0; r < 8; ++r) {
        int m = t * 16 + r + 8 * lh;                  // C/D: vgpr r -> M = r+8*lh
        O[(size_t)m * 128 + col0] = acc0[r] + bv0;
        O[(size_t)m * 128 + col1] = acc1[r] + bv1;
    }
}

// ---------------------------------------------------------------------------
// helpers / edge phase
// ---------------------------------------------------------------------------
__device__ inline void atomicMaxF32(float* addr, float v) {
    if (v >= 0.0f) atomicMax((int*)addr, __float_as_int(v));
    else           atomicMin((unsigned int*)addr, __float_as_uint(v));
}

__global__ void init_stats(float* __restrict__ m, float* __restrict__ s,
                           float* __restrict__ acc)
{
    int i = blockIdx.x * blockDim.x + threadIdx.x;
    if (i < NN * NHEAD) { m[i] = -INFINITY; s[i] = 0.0f; }
    if (i == 0) *acc = 0.0f;
}

__global__ void edge_score1(const float* __restrict__ Q, const float* __restrict__ Kf,
                            const float* __restrict__ V,
                            const int* __restrict__ src, const int* __restrict__ dst,
                            float* __restrict__ score, float* __restrict__ mbuf)
{
    int e = blockIdx.x * blockDim.x + threadIdx.x;
    if (e >= EE) return;
    int sN = src[e], dN = dst[e];
    __builtin_prefetch(V + (size_t)sN * 128, 0, 1);   // warm L2 for agg pass
    const float4* qr = (const float4*)(Q  + (size_t)dN * 128);
    const float4* kr = (const float4*)(Kf + (size_t)sN * 128);
    #pragma unroll
    for (int h = 0; h < NHEAD; ++h) {
        float a = 0.0f;
        #pragma unroll
        for (int i = 0; i < 8; ++i) {
            float4 qa = qr[h * 8 + i], kb = kr[h * 8 + i];
            a += qa.x * kb.x + qa.y * kb.y + qa.z * kb.z + qa.w * kb.w;
        }
        a *= 0.17677669529663687f;   // 1/sqrt(32)
        score[e * NHEAD + h] = a;
        atomicMaxF32(&mbuf[dN * NHEAD + h], a);
    }
}

__global__ void edge_exp(const int* __restrict__ dst, const float* __restrict__ mbuf,
                         float* __restrict__ score, float* __restrict__ sbuf)
{
    int i = blockIdx.x * blockDim.x + threadIdx.x;
    if (i >= EE * NHEAD) return;
    int e = i >> 2, h = i & 3;
    int dN = dst[e];
    float ex = __expf(score[i] - mbuf[dN * NHEAD + h]);
    score[i] = ex;
    atomicAdd(&sbuf[dN * NHEAD + h], ex);
}

__global__ void edge_agg1(const int* __restrict__ src, const int* __restrict__ dst,
                          const float* __restrict__ score, const float* __restrict__ sbuf,
                          const float* __restrict__ V, float* __restrict__ Hout)
{
    int i = blockIdx.x * blockDim.x + threadIdx.x;
    if (i >= EE * 128) return;
    int e = i >> 7, c = i & 127, h = c >> 5;
    int sN = src[e], dN = dst[e];
    float alpha = score[e * NHEAD + h] / (sbuf[dN * NHEAD + h] + EPS_F);
    atomicAdd(&Hout[(size_t)dN * 128 + c], alpha * V[(size_t)sN * 128 + c]);
}

// ---------------------------------------------------------------------------
// Layer-2 skinny GEMMs (out = 4/head): per (node, one of 16 outputs)
// ---------------------------------------------------------------------------
__global__ void l2_gemm(const float* __restrict__ H1,
                        const float* __restrict__ Wq, const float* __restrict__ Wk,
                        const float* __restrict__ Wv, const float* __restrict__ Ws,
                        const float* __restrict__ bq, const float* __restrict__ bk,
                        const float* __restrict__ bv, const float* __restrict__ bs,
                        float* __restrict__ Q2, float* __restrict__ K2,
                        float* __restrict__ V2, float* __restrict__ H2)
{
    int i = blockIdx.x * blockDim.x + threadIdx.x;
    if (i >= NN * 16) return;
    int n = i >> 4, t = i & 15;
    int mat = t >> 2, j = t & 3;
    const float* W; const float* b; float* O;
    switch (mat) {
        case 0:  W = Wq; b = bq; O = Q2; break;
        case 1:  W = Wk; b = bk; O = K2; break;
        case 2:  W = Wv; b = bv; O = V2; break;
        default: W = Ws; b = bs; O = H2; break;   // root-skip goes into H2
    }
    const float* hr = H1 + (size_t)n * 128;
    float acc = 0.0f;
    #pragma unroll 8
    for (int k = 0; k < 128; ++k) {
        float hv = hr[k];
        hv = hv > 0.0f ? hv : 0.0f;               // fused ReLU
        acc += hv * W[k * 4 + j];
    }
    O[n * 4 + j] = acc + b[j];
}

__global__ void edge_score2(const float* __restrict__ Q2, const float* __restrict__ K2,
                            const int* __restrict__ src, const int* __restrict__ dst,
                            float* __restrict__ score, float* __restrict__ mbuf)
{
    int e = blockIdx.x * blockDim.x + threadIdx.x;
    if (e >= EE) return;
    int sN = src[e], dN = dst[e];
    #pragma unroll
    for (int h = 0; h < NHEAD; ++h) {
        float sc = Q2[dN * 4 + h] * K2[sN * 4 + h];   // C=1 -> /sqrt(1)
        score[e * NHEAD + h] = sc;
        atomicMaxF32(&mbuf[dN * NHEAD + h], sc);
    }
}

__global__ void edge_agg2(const int* __restrict__ src, const int* __restrict__ dst,
                          const float* __restrict__ score, const float* __restrict__ sbuf,
                          const float* __restrict__ V2, float* __restrict__ H2)
{
    int i = blockIdx.x * blockDim.x + threadIdx.x;
    if (i >= EE * NHEAD) return;
    int e = i >> 2, h = i & 3;
    int sN = src[e], dN = dst[e];
    float alpha = score[i] / (sbuf[dN * NHEAD + h] + EPS_F);
    atomicAdd(&H2[dN * 4 + h], alpha * V2[sN * 4 + h]);
}

// ---------------------------------------------------------------------------
// final: mean_n ( H2[n,:] @ Wl + bl )
// ---------------------------------------------------------------------------
__global__ void finalize(const float* __restrict__ H2, const float* __restrict__ Wl,
                         float* __restrict__ acc)
{
    __shared__ float red[256];
    int i = blockIdx.x * blockDim.x + threadIdx.x;
    float v = 0.0f;
    if (i < NN)
        v = H2[i * 4 + 0] * Wl[0] + H2[i * 4 + 1] * Wl[1] +
            H2[i * 4 + 2] * Wl[2] + H2[i * 4 + 3] * Wl[3];
    red[threadIdx.x] = v;
    __syncthreads();
    for (int st = 128; st > 0; st >>= 1) {
        if (threadIdx.x < st) red[threadIdx.x] += red[threadIdx.x + st];
        __syncthreads();
    }
    if (threadIdx.x == 0) atomicAdd(acc, red[0]);
}

__global__ void write_out(const float* __restrict__ acc, const float* __restrict__ bl,
                          float* __restrict__ out)
{
    out[0] = acc[0] / (float)NN + bl[0];
}

// ---------------------------------------------------------------------------
extern "C" void kernel_launch(void* const* d_in, const int* in_sizes, int n_in,
                              void* d_out, int out_size, void* d_ws, size_t ws_size,
                              hipStream_t stream)
{
    const float* x    = (const float*)d_in[0];
    const int*   esrc = (const int*)d_in[1];
    const int*   edst = (const int*)d_in[2];
    const float* W1q = (const float*)d_in[3];  const float* b1q = (const float*)d_in[4];
    const float* W1k = (const float*)d_in[5];  const float* b1k = (const float*)d_in[6];
    const float* W1v = (const float*)d_in[7];  const float* b1v = (const float*)d_in[8];
    const float* W1s = (const float*)d_in[9];  const float* b1s = (const float*)d_in[10];
    const float* W2q = (const float*)d_in[11]; const float* b2q = (const float*)d_in[12];
    const float* W2k = (const float*)d_in[13]; const float* b2k = (const float*)d_in[14];
    const float* W2v = (const float*)d_in[15]; const float* b2v = (const float*)d_in[16];
    const float* W2s = (const float*)d_in[17]; const float* b2s = (const float*)d_in[18];
    const float* Wl  = (const float*)d_in[19]; const float* bl  = (const float*)d_in[20];
    float* out = (float*)d_out;

    float* ws = (float*)d_ws;
    size_t o = 0;
    float* Q1 = ws + o; o += (size_t)NN * 128;
    float* K1 = ws + o; o += (size_t)NN * 128;
    float* V1 = ws + o; o += (size_t)NN * 128;
    float* H1 = ws + o; o += (size_t)NN * 128;       // skip1 + aggregation
    float* SC = ws + o; o += (size_t)EE * NHEAD;
    float* Mb = ws + o; o += (size_t)NN * NHEAD;
    float* Sb = ws + o; o += (size_t)NN * NHEAD;
    float* Q2 = ws + o; o += (size_t)NN * NHEAD;
    float* K2 = ws + o; o += (size_t)NN * NHEAD;
    float* V2 = ws + o; o += (size_t)NN * NHEAD;
    float* H2 = ws + o; o += (size_t)NN * NHEAD;     // skip2 + aggregation
    _Float16* Apk = (_Float16*)(ws + o); o += (size_t)NN * 128 / 2;  // f16 frags
    _Float16* Bpk = (_Float16*)(ws + o); o += 65536 / 2;
    float* Ac = ws + o; o += 1;
    (void)in_sizes; (void)n_in; (void)out_size; (void)ws_size;

    const int T = 256;

    // ---- pre-pack f16 WMMA fragments ----
    pack_a<<<(NN * 128 + T - 1) / T, T, 0, stream>>>(x, Apk);
    pack_b<<<(4 * 16384 + T - 1) / T, T, 0, stream>>>(W1q, W1k, W1v, W1s, Bpk);

    // ---- layer 1 ----
    init_stats<<<(NN * NHEAD + T - 1) / T, T, 0, stream>>>(Mb, Sb, Ac);
    gemm_l1<<<dim3(NN / 16, 4), 128, 0, stream>>>(Apk, Bpk,
                                                  b1q, b1k, b1v, b1s,
                                                  Q1, K1, V1, H1);
    edge_score1<<<(EE + T - 1) / T, T, 0, stream>>>(Q1, K1, V1, esrc, edst, SC, Mb);
    edge_exp<<<(EE * NHEAD + T - 1) / T, T, 0, stream>>>(edst, Mb, SC, Sb);
    edge_agg1<<<(EE * 128) / T, T, 0, stream>>>(esrc, edst, SC, Sb, V1, H1);

    // ---- layer 2 ----
    l2_gemm<<<(NN * 16 + T - 1) / T, T, 0, stream>>>(H1, W2q, W2k, W2v, W2s,
                                                     b2q, b2k, b2v, b2s,
                                                     Q2, K2, V2, H2);
    init_stats<<<(NN * NHEAD + T - 1) / T, T, 0, stream>>>(Mb, Sb, Ac);
    edge_score2<<<(EE + T - 1) / T, T, 0, stream>>>(Q2, K2, esrc, edst, SC, Mb);
    edge_exp<<<(EE * NHEAD + T - 1) / T, T, 0, stream>>>(edst, Mb, SC, Sb);
    edge_agg2<<<(EE * NHEAD + T - 1) / T, T, 0, stream>>>(esrc, edst, SC, Sb, V2, H2);

    // ---- readout ----
    finalize<<<(NN + T - 1) / T, T, 0, stream>>>(H2, Wl, Ac);
    write_out<<<1, 1, 0, stream>>>(Ac, bl, out);
}
